// ContrastivePatchLoss_41248865911083
// MI455X (gfx1250) — compile-verified, matrix operand order checked by reading, not run on hardware
//
#include <hip/hip_runtime.h>
#include <hip/hip_bf16.h>

typedef __attribute__((ext_vector_type(16))) _Float16 v16h;
typedef __attribute__((ext_vector_type(8)))  _Float16 v8h;
typedef __attribute__((ext_vector_type(8)))  float    v8f;
typedef __attribute__((ext_vector_type(4)))  unsigned int u32x4;
typedef __attribute__((ext_vector_type(8)))  int      i32x8;
typedef __attribute__((ext_vector_type(4)))  int      i32x4;

union Frag { v16h v; v8h h[2]; };

#define NPATCH 512      // B * PATCH * PATCH
#define HW     64       // rows per patch
#define CDIM   256
#define MROWS  2048     // bank rows (L * h * w)
#define SUPER  64       // bank rows per TDM super-tile
#define NSUPER 32       // MROWS / SUPER
#define INV_TEMP 2.0f
#define LOG2E  1.4426950408889634f
#define LN2    0.6931471805599453f
#define SCALE2 (INV_TEMP * LOG2E)   // folded into anchor f16 conversion
#define EPSV   1e-5f

__device__ __forceinline__ float fexp2(float x) { return __builtin_amdgcn_exp2f(x); }

// ------- preprocessing: [B,C,H,W] -> f16 patches [512,64,256], pre-scaled ---
__global__ void k_to_patches(const float* __restrict__ in, _Float16* __restrict__ out,
                             float scale) {
    int i = blockIdx.x * blockDim.x + threadIdx.x;       // 8,388,608 elements
    int c  = i & 255;
    int n  = (i >> 8) & 63;
    int p  = i >> 14;
    int b  = p >> 6;
    int j  = (p >> 3) & 7;
    int kk = p & 7;
    int y  = n >> 3;
    int x  = n & 7;
    size_t off = (((size_t)b * CDIM + c) * 64 + (j * 8 + y)) * 64 + (kk * 8 + x);
    out[i] = (_Float16)(in[off] * scale);
}

// ---------------- banks: [L,C,8,8] -> f16 [2048,256] ------------------------
__global__ void k_bank(const float* __restrict__ in, _Float16* __restrict__ out) {
    int i = blockIdx.x * blockDim.x + threadIdx.x;       // 524,288 elements
    int c = i & 255;
    int m = i >> 8;
    int l = m >> 6;
    int y = (m >> 3) & 7;
    int x = m & 7;
    size_t off = (((size_t)l * CDIM + c) * 8 + y) * 8 + x;
    out[i] = (_Float16)in[off];
}

// ---------------- per-patch label mean -> use_pos ---------------------------
__global__ void k_lab(const float* __restrict__ lab, int* __restrict__ use_pos) {
    int p = blockIdx.x;                 // 512 blocks, 256 threads
    int b = p >> 6, j = (p >> 3) & 7, k = p & 7;
    int t = threadIdx.x;
    float s = 0.f;
    #pragma unroll
    for (int e = 0; e < 4; ++e) {
        int idx = e * 256 + t;          // 0..1023 over the 32x32 block
        int y = idx >> 5, x = idx & 31;
        s += lab[((size_t)b * 256 + (j * 32 + y)) * 256 + (k * 32 + x)];
    }
    for (int off = 16; off; off >>= 1) s += __shfl_down(s, off);
    __shared__ float wsum[8];
    if ((t & 31) == 0) wsum[t >> 5] = s;
    __syncthreads();
    if (t == 0) {
        float tot = 0.f;
        #pragma unroll
        for (int w = 0; w < 8; ++w) tot += wsum[w];
        use_pos[p] = (tot * (1.0f / 1024.0f)) < 0.1f ? 1 : 0;
    }
}

// ------- pos_sim2[row] = dot(scaled_anchor, pos) == pos_sim * log2e ---------
__global__ void k_possim(const _Float16* __restrict__ a, const _Float16* __restrict__ b,
                         float* __restrict__ out) {
    int wid = threadIdx.x >> 5, lane = threadIdx.x & 31;
    int row = blockIdx.x * 8 + wid;                         // 32768 rows
    const v8h* pa = (const v8h*)(a + (size_t)row * CDIM + lane * 8);
    const v8h* pb = (const v8h*)(b + (size_t)row * CDIM + lane * 8);
    v8h va = *pa, vb = *pb;
    float acc = 0.f;
    #pragma unroll
    for (int e = 0; e < 8; ++e) acc += (float)va[e] * (float)vb[e];
    for (int off = 16; off; off >>= 1) acc += __shfl_down(acc, off);
    if (lane == 0) out[row] = acc;      // already log2-domain (anchor scaled)
}

// ---------------- TDM: DMA one 64x256 f16 super-tile (32 KB) into LDS -------
// D# per CDNA5 ISA:
//   group0: count=1 | lds_addr | global_addr[56:0] | type=2
//   group1: data_size=1(2B); tensor_dim0=256; tensor_dim1=64;
//           tile_dim0=256; tile_dim1=64; tensor_dim0_stride=256
__device__ __forceinline__ void tdm_load_tile(const _Float16* gsrc, unsigned lds_off) {
    unsigned long long ga = (unsigned long long)(uintptr_t)gsrc;
    u32x4 g0;
    g0[0] = 1u;                                                // count=1
    g0[1] = lds_off;                                           // lds byte addr
    g0[2] = (unsigned)(ga & 0xffffffffu);                      // global_addr lo
    g0[3] = (unsigned)((ga >> 32) & 0x01ffffffu) | (2u << 30); // addr hi | type=2
    i32x8 g1;
    g1[0] = (int)(1u << 16);        // data_size=1 (2 bytes), mask=0
    g1[1] = (int)(256u << 16);      // tensor_dim0[15:0]=256 at bits[63:48]
    g1[2] = (int)(64u << 16);       // tensor_dim1[15:0]=64  at bits[95:80]
    g1[3] = (int)(256u << 16);      // tile_dim0=256 at bits[127:112]
    g1[4] = 64;                     // tile_dim1=64 at bits[143:128]
    g1[5] = 256;                    // tensor_dim0_stride lo32 = 256
    g1[6] = 0;
    g1[7] = 0;
    i32x4 z4 = {};
    i32x8 z8 = {};
    __builtin_amdgcn_tensor_load_to_lds(g0, g1, z4, z4, z8, 0);
}

// ---------------- fragment load helper --------------------------------------
__device__ __forceinline__ void loadFrag(Frag& f, const _Float16* base, int ks) {
    f.h[0] = *(const v8h*)(base + ks * 32);
    f.h[1] = *(const v8h*)(base + ks * 32 + 16);
}

// -------- super-tile compute: 4 interleaved WMMA chains + merged softmax ----
// The 4 sub-tiles differ only in N, so all feed the same 8 row-sums: one
// 32-way max, one rescale, 32 exps per super-tile.
__device__ __forceinline__ void superUpdate(const Frag (&a)[8], const _Float16* lb,
                                            int hrow, int hi, float& m, float (&s)[8]) {
    const _Float16* base = lb + hrow * CDIM + hi * 8;
    v8f acc[4] = {};
    #pragma unroll
    for (int ks = 0; ks < 8; ++ks) {
        Frag b[4];
        #pragma unroll
        for (int c = 0; c < 4; ++c) loadFrag(b[c], base + c * 16 * CDIM, ks);
        #pragma unroll
        for (int c = 0; c < 4; ++c) {
            acc[c] = __builtin_amdgcn_wmma_f32_16x16x32_f16(
                false, a[ks].v, false, b[c].v, (short)0, acc[c], false, false);
        }
    }
    // acc values are sims in log2 units (scale folded into A)
    float t = acc[0][0];
    #pragma unroll
    for (int c = 0; c < 4; ++c) {
        #pragma unroll
        for (int r = 0; r < 8; ++r) t = fmaxf(t, acc[c][r]);
    }
    if (t > m) {                                             // rare after warmup
        float f = fexp2(m - t);
        #pragma unroll
        for (int r = 0; r < 8; ++r) s[r] *= f;
        m = t;
    }
    #pragma unroll
    for (int c = 0; c < 4; ++c) {
        #pragma unroll
        for (int r = 0; r < 8; ++r) s[r] += fexp2(acc[c][r] - m);
    }
}

// ---------------- main: one block per patch; wave = M-tile; B via TDM+LDS ---
__global__ __launch_bounds__(128)
void k_main(const _Float16* __restrict__ anchors, const _Float16* __restrict__ negb,
            const _Float16* __restrict__ posb, const float* __restrict__ possim,
            const int* __restrict__ usepos, float* __restrict__ out) {
    const int p    = blockIdx.x;           // 512 blocks
    const int wid  = threadIdx.x >> 5;     // wave = M-tile (4 per patch)
    const int lane = threadIdx.x & 31;
    const int hrow = lane & 15;
    const int hi   = lane >> 4;

    __shared__ _Float16 lbuf[2][SUPER * CDIM];     // 2 x 32 KB B super-tiles

    // A fragments: this wave's 16 rows, K=256, held in VGPRs
    const _Float16* Abase =
        anchors + (((size_t)p * HW) + wid * 16 + hrow) * CDIM + hi * 8;
    Frag a[8];
    #pragma unroll
    for (int ks = 0; ks < 8; ++ks) loadFrag(a[ks], Abase, ks);

    int up = __builtin_amdgcn_readfirstlane(usepos[p]);   // keep D# scalar
    const _Float16* bank = up ? posb : negb;

    unsigned lds0 = (unsigned)(uintptr_t)&lbuf[0][0];
    unsigned lds1 = (unsigned)(uintptr_t)&lbuf[1][0];

    if (wid == 0) tdm_load_tile(bank, lds0);

    float m = -1e30f;
    float s[8];
    #pragma unroll
    for (int r = 0; r < 8; ++r) s[r] = 0.f;

    #pragma unroll 1
    for (int i = 0; i < NSUPER; ++i) {
        if (wid == 0) __builtin_amdgcn_s_wait_tensorcnt(0);   // super-tile i in
        __syncthreads();                                      // publish to all
        if (wid == 0 && i + 1 < NSUPER)                       // prefetch i+1
            tdm_load_tile(bank + (size_t)(i + 1) * SUPER * CDIM,
                          (i & 1) ? lds0 : lds1);
        superUpdate(a, &lbuf[i & 1][0], hrow, hi, m, s);
    }

    // merge (m, s[0..7]) across the 16-lane half-groups (xor <= 8 in-group);
    // shared m -> only 2 exps per step
    #pragma unroll
    for (int off = 1; off < 16; off <<= 1) {
        float mo = __shfl_xor(m, off);
        float mn = fmaxf(m, mo);
        float fs = fexp2(m - mn);
        float fo = fexp2(mo - mn);
        #pragma unroll
        for (int r = 0; r < 8; ++r) {
            float so = __shfl_xor(s[r], off);
            s[r] = fmaf(so, fo, s[r] * fs);
        }
        m = mn;
    }

    // each wave's rows are complete over all N: fold in pos term, emit loss
    float part = 0.f;
    if (hrow == 0) {                       // lanes 0 (rows +0..7), 16 (+8..15)
        #pragma unroll
        for (int r = 0; r < 8; ++r) {
            int row = wid * 16 + hi * 8 + r;
            float pos2 = possim[(size_t)p * HW + row];       // log2-domain
            float mt  = fmaxf(m, pos2);
            float ep  = fexp2(pos2 - mt);
            float den = s[r] * fexp2(m - mt) + ep + EPSV;
            part += -LN2 * __builtin_amdgcn_logf(ep / den + EPSV);
        }
    }
    part += __shfl_xor(part, 16);
    if (lane == 0) atomicAdd(out, part * (1.0f / (NPATCH * HW)));
}

extern "C" void kernel_launch(void* const* d_in, const int* in_sizes, int n_in,
                              void* d_out, int out_size, void* d_ws, size_t ws_size,
                              hipStream_t stream) {
    (void)in_sizes; (void)n_in; (void)out_size; (void)ws_size;
    const float* main_out   = (const float*)d_in[0];
    const float* ema_out    = (const float*)d_in[1];
    const float* main_label = (const float*)d_in[2];
    const float* neg_banks  = (const float*)d_in[3];
    const float* pos_banks  = (const float*)d_in[4];
    float* out = (float*)d_out;

    // workspace carve-up (bytes)
    char* ws = (char*)d_ws;
    _Float16* anchors_h = (_Float16*)(ws);                               // 16.78 MB
    _Float16* pospair_h = (_Float16*)(ws + 16777216);                    // 16.78 MB
    _Float16* neg_h     = (_Float16*)(ws + 33554432);                    // 1 MB
    _Float16* pos_h     = (_Float16*)(ws + 34603008);                    // 1 MB
    float*    possim    = (float*)   (ws + 35651584);                    // 128 KB
    int*      usepos    = (int*)     (ws + 35782656);                    // 2 KB

    (void)hipMemsetAsync(out, 0, sizeof(float), stream);

    const int PT = NPATCH * HW * CDIM;   // 8,388,608
    k_to_patches<<<PT / 256, 256, 0, stream>>>(main_out, anchors_h, SCALE2);
    k_to_patches<<<PT / 256, 256, 0, stream>>>(ema_out,  pospair_h, 1.0f);

    const int BT = MROWS * CDIM;         // 524,288
    k_bank<<<BT / 256, 256, 0, stream>>>(neg_banks, neg_h);
    k_bank<<<BT / 256, 256, 0, stream>>>(pos_banks, pos_h);

    k_lab<<<NPATCH, 256, 0, stream>>>(main_label, usepos);

    k_possim<<<(NPATCH * HW) / 8, 256, 0, stream>>>(anchors_h, pospair_h, possim);

    k_main<<<NPATCH, 128, 0, stream>>>(anchors_h, neg_h, pos_h, possim, usepos, out);
}
